// MyLSTM_21234318311800
// MI455X (gfx1250) — matrix operand order repaired and checked by reading, not compiled, not run on hardware
//
#include <hip/hip_runtime.h>
#include <hip/hip_bf16.h>
#include <math.h>

typedef unsigned short ushort_t;
typedef __attribute__((ext_vector_type(16))) __bf16        v16bf;
typedef __attribute__((ext_vector_type(8)))  float         v8f;
typedef __attribute__((ext_vector_type(8)))  unsigned int  v8u;

#define T_SEQ 2048
#define E_DIM 512
#define H_DIM 1024
#define G4    4096      // 4*H
#define OUTN  30000
#define BI    2048      // 2*H

__device__ __forceinline__ ushort_t f2bf(float f) {
  unsigned u = __float_as_uint(f);
  unsigned r = u + 0x7fffu + ((u >> 16) & 1u);   // round-to-nearest-even
  return (ushort_t)(r >> 16);
}
__device__ __forceinline__ float bflo(unsigned u) { return __uint_as_float(u << 16); }
__device__ __forceinline__ float bfhi(unsigned u) { return __uint_as_float(u & 0xffff0000u); }

// ---------------------------------------------------------------------------
// fp32 -> bf16 bulk convert (grid-stride)
// ---------------------------------------------------------------------------
__global__ void cvt_f32_bf16(const float* __restrict__ in, ushort_t* __restrict__ out, int n) {
  int stride = blockDim.x * gridDim.x;
  for (int i = blockIdx.x * blockDim.x + threadIdx.x; i < n; i += stride)
    out[i] = f2bf(in[i]);
}

// ---------------------------------------------------------------------------
// C[M,N] (f32) = A[M,K](bf16,row-major) * B[N,K](bf16,row-major)^T + bias[N]
// Block tile 128(M) x 128(N), BK=32, 8 waves: wave w -> rows [16w,16w+16),
// all 128 N (8 accumulators). B fragments are preloaded as a batch so the
// ds_load_b128s group under one s_wait_dscnt and the 8 v_wmma issue
// back-to-back (no D->A/B hazards, only D->C accumulation chains).
// ---------------------------------------------------------------------------
__global__ __launch_bounds__(256) void gemm_bf16(
    const ushort_t* __restrict__ A, const ushort_t* __restrict__ B,
    const float* __restrict__ bias, float* __restrict__ C,
    int M, int N, int K, int ldc)
{
  __shared__ __align__(16) ushort_t As[128][40];  // padded stride 80B (16B-aligned)
  __shared__ __align__(16) ushort_t Bs[128][40];

  const int tid  = threadIdx.x;
  const int m0   = blockIdx.y * 128;
  const int n0   = blockIdx.x * 128;
  const int wv   = tid >> 5;          // wave 0..7 (wave32)
  const int lane = tid & 31;
  const int half = lane >> 4;
  const int mf   = lane & 15;

  v8f acc[8] = {};

  const int KT = K >> 5;
  for (int kt = 0; kt < KT; ++kt) {
    const int ko = kt << 5;
    __syncthreads();
    // A tile: 128 rows x 32 cols, 16B chunks (512 chunks, 2 per thread)
    #pragma unroll
    for (int i = 0; i < 2; ++i) {
      int idx = tid + i * 256;
      int r = idx >> 2, c = idx & 3;
      uint4 v = *(const uint4*)&A[(size_t)(m0 + r) * K + ko + c * 8];
      *(uint4*)&As[r][c * 8] = v;
    }
    // B tile: 128 rows x 32 cols (512 chunks, 2 per thread), guard n >= N
    #pragma unroll
    for (int i = 0; i < 2; ++i) {
      int idx = tid + i * 256;
      int r = idx >> 2, c = idx & 3;
      int n = n0 + r;
      uint4 v = make_uint4(0u, 0u, 0u, 0u);
      if (n < N) v = *(const uint4*)&B[(size_t)n * K + ko + c * 8];
      *(uint4*)&Bs[r][c * 8] = v;
    }
    if (kt + 1 < KT)  // CDNA5 global_prefetch_b8 of next A k-slice
      __builtin_prefetch(&A[(size_t)(m0 + (tid >> 1)) * K + ko + 32], 0, 0);
    __syncthreads();

    // A fragment, 16-bit A 16x32 layout: lanes0-15 V0..3:K=0..7,V4..7:K=16..23;
    // lanes16-31: +8. Two contiguous 16B runs -> two ds_load_b128.
    v8u au;
    #pragma unroll
    for (int v = 0; v < 8; ++v) {
      int kb = (v < 4 ? 2 * v : 16 + 2 * (v - 4)) + 8 * half;
      au[v] = *(const unsigned*)&As[wv * 16 + mf][kb];
    }
    v16bf af = __builtin_bit_cast(v16bf, au);

    // Preload all 8 B fragments (32x16 layout: n = lane&15; same K order as A:
    // lanes0-15 K=0..7,16..23; lanes16-31 K=8..15,24..31).
    v16bf bfrag[8];
    #pragma unroll
    for (int nt = 0; nt < 8; ++nt) {
      v8u bu;
      #pragma unroll
      for (int i = 0; i < 8; ++i)
        bu[i] = *(const unsigned*)&Bs[nt * 16 + mf][(i < 4 ? 2 * i : 16 + 2 * (i - 4)) + 8 * half];
      bfrag[nt] = __builtin_bit_cast(v16bf, bu);
    }
    #pragma unroll
    for (int nt = 0; nt < 8; ++nt)
      acc[nt] = __builtin_amdgcn_wmma_f32_16x16x32_bf16(
          false, af, false, bfrag[nt], (short)0, acc[nt], false, false);
  }

  // Epilogue: D layout — VGPR r: lanes0-15 -> M=r, lanes16-31 -> M=8+r; N=lane&15
  #pragma unroll
  for (int nt = 0; nt < 8; ++nt) {
    int ng = n0 + nt * 16 + mf;
    if (ng >= N) continue;
    float bb = bias ? bias[ng] : 0.0f;
    #pragma unroll
    for (int r = 0; r < 8; ++r) {
      int mg = m0 + wv * 16 + r + (half ? 8 : 0);
      C[(size_t)mg * ldc + ng] = acc[nt][r] + bb;
    }
  }
}

// ---------------------------------------------------------------------------
// Persistent bidirectional LSTM recurrence.
// 64 blocks: dir = blkIdx>>5 (fwd/bwd), 32 blocks/dir x 32 hidden units each.
// Per step: stage h (ping-pong global buf) in LDS, 256 threads = 128 rows x
// 2 K-halves of bf16 GEMV against L2-resident W_hh, pointwise gate math,
// device-scope atomic-counter grid barrier (monotonic; reset by init kernel).
// ---------------------------------------------------------------------------
__global__ __launch_bounds__(256) void lstm_seq(
    const ushort_t* __restrict__ Whh,   // [2][4096][1024] bf16
    const float*    __restrict__ xw,    // [2][T][4096] fp32
    float* hbuf,                        // [2 bufs][2 dir][1024]
    unsigned* bar,
    ushort_t* __restrict__ bi_bf)       // [T][2048] bf16 (h_fwd | h_bwd)
{
  __shared__ float sh_h[1024];
  __shared__ float part[256];
  __shared__ float gl[128];

  const int tid  = threadIdx.x;
  const int dir  = blockIdx.x >> 5;
  const int blk  = blockIdx.x & 31;
  const int ksel = tid >> 7;        // K half: [0,512) or [512,1024)
  const int rl   = tid & 127;       // local row: gate*32 + unit
  const int gate = rl >> 5;
  const int ul   = rl & 31;
  const int u    = blk * 32 + ul;
  const int row  = gate * 1024 + u; // PyTorch gate order i,f,g,o

  const ushort_t* wrow = Whh + (size_t)dir * G4 * H_DIM + (size_t)row * H_DIM + ksel * 512;
  const float*    xwd  = xw + (size_t)dir * T_SEQ * G4;
  const unsigned  nb   = gridDim.x;
  float c_reg = 0.0f;               // cell state lives in tid<32 threads

  for (int s = 0; s < T_SEQ; ++s) {
    const int t = dir ? (T_SEQ - 1 - s) : s;
    const float* hsrc = hbuf + (((s & 1) * 2 + dir) * 1024);
    float*       hdst = hbuf + ((((s + 1) & 1) * 2 + dir) * 1024);

    for (int k = tid; k < 1024; k += 256)
      sh_h[k] = __hip_atomic_load(&hsrc[k], __ATOMIC_RELAXED, __HIP_MEMORY_SCOPE_AGENT);
    __syncthreads();

    float acc = 0.0f;
    const uint4* wp = (const uint4*)wrow;     // 64 x (8 bf16)
    const float* hp = &sh_h[ksel * 512];
    #pragma unroll 4
    for (int j = 0; j < 64; ++j) {
      uint4 w8 = wp[j];
      const float* h8 = hp + j * 8;
      acc += h8[0] * bflo(w8.x) + h8[1] * bfhi(w8.x)
           + h8[2] * bflo(w8.y) + h8[3] * bfhi(w8.y)
           + h8[4] * bflo(w8.z) + h8[5] * bfhi(w8.z)
           + h8[6] * bflo(w8.w) + h8[7] * bfhi(w8.w);
    }
    part[tid] = acc;
    __syncthreads();
    if (tid < 128)
      gl[tid] = part[tid] + part[tid + 128]
              + xwd[(size_t)t * G4 + (tid >> 5) * 1024 + blk * 32 + (tid & 31)];
    __syncthreads();
    if (tid < 32) {
      float gi = gl[tid], gf = gl[32 + tid], gg = gl[64 + tid], go = gl[96 + tid];
      float i = 1.0f / (1.0f + __expf(-gi));
      float f = 1.0f / (1.0f + __expf(-gf));
      float g = tanhf(gg);
      float o = 1.0f / (1.0f + __expf(-go));
      c_reg = f * c_reg + i * g;
      float h = o * tanhf(c_reg);
      __hip_atomic_store(&hdst[blk * 32 + tid], h, __ATOMIC_RELAXED, __HIP_MEMORY_SCOPE_AGENT);
      bi_bf[(size_t)t * BI + dir * H_DIM + blk * 32 + tid] = f2bf(h);
    }
    __threadfence();
    __syncthreads();
    if (tid == 0) {
      __hip_atomic_fetch_add(bar, 1u, __ATOMIC_ACQ_REL, __HIP_MEMORY_SCOPE_AGENT);
      const unsigned tgt = (unsigned)(s + 1) * nb;
      while (__hip_atomic_load(bar, __ATOMIC_ACQUIRE, __HIP_MEMORY_SCOPE_AGENT) < tgt)
        __builtin_amdgcn_s_sleep(1);
    }
    __syncthreads();
  }
}

// ---------------------------------------------------------------------------
// In-place row-wise log_softmax: one block per row, online max/logsumexp.
// ---------------------------------------------------------------------------
__global__ __launch_bounds__(256) void log_softmax_rows(float* __restrict__ out, int n) {
  __shared__ float sm[256], ss[256];
  float* row = out + (size_t)blockIdx.x * n;
  const int tid = threadIdx.x;
  float m = -INFINITY, s = 0.0f;
  for (int k = tid; k < n; k += 256) {
    float x = row[k];
    if (x > m) { s = s * __expf(m - x) + 1.0f; m = x; }
    else       { s += __expf(x - m); }
  }
  sm[tid] = m; ss[tid] = s;
  __syncthreads();
  for (int o = 128; o > 0; o >>= 1) {
    if (tid < o) {
      float m2 = sm[tid + o], s2 = ss[tid + o];
      float mm = fmaxf(sm[tid], m2);
      ss[tid] = ss[tid] * __expf(sm[tid] - mm) + s2 * __expf(m2 - mm);
      sm[tid] = mm;
    }
    __syncthreads();
  }
  const float lse = sm[0] + __logf(ss[0]);
  __syncthreads();
  for (int k = tid; k < n; k += 256) row[k] -= lse;
}

// ---------------------------------------------------------------------------
// Per-call state reset (barrier counter + h ping-pong buffers) — keeps the
// persistent kernel deterministic across graph replays.
// ---------------------------------------------------------------------------
__global__ void init_state(unsigned* bar, float* hbuf) {
  int i = blockIdx.x * blockDim.x + threadIdx.x;
  if (i < 16) bar[i] = 0u;
  for (int k = i; k < 4096; k += blockDim.x * gridDim.x) hbuf[k] = 0.0f;
}

extern "C" void kernel_launch(void* const* d_in, const int* in_sizes, int n_in,
                              void* d_out, int out_size, void* d_ws, size_t ws_size,
                              hipStream_t stream) {
  (void)in_sizes; (void)n_in; (void)out_size; (void)ws_size;
  const float* embeds = (const float*)d_in[0];
  const float* W_ih_f = (const float*)d_in[1];
  const float* W_hh_f = (const float*)d_in[2];
  const float* b_f    = (const float*)d_in[3];
  const float* W_ih_b = (const float*)d_in[4];
  const float* W_hh_b = (const float*)d_in[5];
  const float* b_b    = (const float*)d_in[6];
  const float* W_fc   = (const float*)d_in[7];
  const float* b_fc   = (const float*)d_in[8];
  float* out = (float*)d_out;

  // ---- workspace carve-up (~226 MB) ----
  char* w = (char*)d_ws;
  unsigned* bar  = (unsigned*)w;                       // 64 B
  float*    hbuf = (float*)(w + 64);                   // 4096 f32 ping-pong h
  size_t off = 64 + 4096 * sizeof(float);
  ushort_t* emb_bf = (ushort_t*)(w + off); off += (size_t)T_SEQ * E_DIM * 2;
  ushort_t* wih_bf = (ushort_t*)(w + off); off += (size_t)2 * G4 * E_DIM * 2;
  ushort_t* whh_bf = (ushort_t*)(w + off); off += (size_t)2 * G4 * H_DIM * 2;
  ushort_t* wfc_bf = (ushort_t*)(w + off); off += (size_t)OUTN * BI * 2;
  float*    xw     = (float*)(w + off);    off += (size_t)2 * T_SEQ * G4 * sizeof(float);
  ushort_t* bi_bf  = (ushort_t*)(w + off); off += (size_t)T_SEQ * BI * 2;

  init_state<<<1, 256, 0, stream>>>(bar, hbuf);

  // fp32 -> bf16 staging (weights stay resident in 192MB L2 afterwards)
  cvt_f32_bf16<<<1024, 256, 0, stream>>>(embeds, emb_bf, T_SEQ * E_DIM);
  cvt_f32_bf16<<<2048, 256, 0, stream>>>(W_ih_f, wih_bf,                    G4 * E_DIM);
  cvt_f32_bf16<<<2048, 256, 0, stream>>>(W_ih_b, wih_bf + (size_t)G4*E_DIM, G4 * E_DIM);
  cvt_f32_bf16<<<4096, 256, 0, stream>>>(W_hh_f, whh_bf,                    G4 * H_DIM);
  cvt_f32_bf16<<<4096, 256, 0, stream>>>(W_hh_b, whh_bf + (size_t)G4*H_DIM, G4 * H_DIM);
  cvt_f32_bf16<<<8192, 256, 0, stream>>>(W_fc,   wfc_bf, OUTN * BI);

  // xW = embeds @ W_ih^T + b, both directions   [2048,4096] each, WMMA bf16
  gemm_bf16<<<dim3(G4 / 128, T_SEQ / 128), 256, 0, stream>>>(
      emb_bf, wih_bf, b_f, xw, T_SEQ, G4, E_DIM, G4);
  gemm_bf16<<<dim3(G4 / 128, T_SEQ / 128), 256, 0, stream>>>(
      emb_bf, wih_bf + (size_t)G4 * E_DIM, b_b,
      xw + (size_t)T_SEQ * G4, T_SEQ, G4, E_DIM, G4);

  // sequential recurrence, both directions concurrently (persistent grid)
  lstm_seq<<<64, 256, 0, stream>>>(whh_bf, xw, hbuf, bar, bi_bf);

  // logits = bi @ W_fc^T + b_fc   [2048,30000], WMMA bf16 (dominant GEMM)
  gemm_bf16<<<dim3((OUTN + 127) / 128, T_SEQ / 128), 256, 0, stream>>>(
      bi_bf, wfc_bf, b_fc, out, T_SEQ, OUTN, BI, OUTN);

  // log_softmax in place on d_out
  log_softmax_rows<<<T_SEQ, 256, 0, stream>>>(out, OUTN);
}
